// GemmRSBaseline_79920751444179
// MI455X (gfx1250) — compile-verified, matrix operand order checked by reading, not run on hardware
//
#include <hip/hip_runtime.h>

// ---------------------------------------------------------------------------
// reduce-scatter GEMM == one dense 8192x8192x8192 fp32 GEMM:
//   out[m, n] = sum_{r,k} x[r, m, k] * w[r, k, n]
// bf16 WMMA (v_wmma_f32_16x16x32_bf16) with 2-term hi/lo operand split
// (3 WMMA passes -> ~fp32 accuracy at 8x the MAC density of f32 WMMA K=4).
// Block tile 128(M) x 256(N), 8 wave32s, wave tile 64x64 (16 accumulators)
// -> 48 WMMAs per 32 fragment ds_load_b128 per step (LDS-bandwidth relief
//    vs. the 1:1 ratio of the 64x32 wave tile).
// ---------------------------------------------------------------------------

typedef __bf16 bf16_t;
typedef __attribute__((ext_vector_type(16))) __bf16 v16bf;
typedef __attribute__((ext_vector_type(8)))  __bf16 v8bf;
typedef __attribute__((ext_vector_type(2)))  __bf16 v2bf;
typedef __attribute__((ext_vector_type(8)))  float  v8f;

#define M_DIM   8192
#define N_DIM   8192
#define LK      1024
#define RANKS   8
#define BTM     128            // block tile rows (M)
#define BTN     256            // block tile cols (N)
#define KT      32             // k slab per step (matches bf16 WMMA K)
#define LDSS    40             // padded LDS row stride in bf16 elems (80 B)
#define STEPS   (RANKS * (LK / KT))   // 256

__global__ __launch_bounds__(256)
void gemm_rs_wmma_bf16x3(const float* __restrict__ X,
                         const float* __restrict__ W,
                         float* __restrict__ Out) {
  __shared__ bf16_t Ahi[BTM * LDSS];
  __shared__ bf16_t Alo[BTM * LDSS];
  __shared__ bf16_t Bhi[BTN * LDSS];   // stored transposed: [n][k]
  __shared__ bf16_t Blo[BTN * LDSS];

  const int tid  = threadIdx.x;
  const int lane = tid & 31;
  const int wid  = tid >> 5;
  const int wm   = wid >> 2;            // 0..1  -> 64-row slab
  const int wn   = wid & 3;             // 0..3  -> 64-col slab
  const int m0   = blockIdx.y * BTM;
  const int n0   = blockIdx.x * BTN;

  // A staging: thread -> (row 0..127, 16 contiguous k)
  const int a_row = tid >> 1;
  const int a_col = (tid & 1) * 16;
  // B staging: thread -> (k pair, 16 contiguous n)
  const int b_k = (tid >> 4) * 2;       // 0,2,...,30
  const int b_n = (tid & 15) * 16;      // 0..240

  v8f acc[4][4];
#pragma unroll
  for (int ms = 0; ms < 4; ++ms)
#pragma unroll
    for (int ns = 0; ns < 4; ++ns)
      acc[ms][ns] = (v8f)0.0f;

  float4 a4[4];                         // 16 floats of A
  float4 b4[8];                         // 32 floats of B (2 k-rows x 16 n)

  auto load_step = [&](int s) {
    const int r  = s >> 5;
    const int k0 = (s & 31) * KT;
    const float* ap = X + (size_t)r * M_DIM * LK
                        + (size_t)(m0 + a_row) * LK + (k0 + a_col);
    const float4* ap4 = reinterpret_cast<const float4*>(ap);
    a4[0] = ap4[0]; a4[1] = ap4[1]; a4[2] = ap4[2]; a4[3] = ap4[3];

    const float* bp = W + (size_t)r * LK * N_DIM
                        + (size_t)(k0 + b_k) * N_DIM + (n0 + b_n);
    const float4* bp0 = reinterpret_cast<const float4*>(bp);
    const float4* bp1 = reinterpret_cast<const float4*>(bp + N_DIM);
    b4[0] = bp0[0]; b4[1] = bp0[1]; b4[2] = bp0[2]; b4[3] = bp0[3];
    b4[4] = bp1[0]; b4[5] = bp1[1]; b4[6] = bp1[2]; b4[7] = bp1[3];
  };

  auto prefetch_step = [&](int s) {
    const int r  = s >> 5;
    const int k0 = (s & 31) * KT;
    const float* ap = X + (size_t)r * M_DIM * LK
                        + (size_t)(m0 + a_row) * LK + (k0 + a_col);
    const float* bp = W + (size_t)r * LK * N_DIM
                        + (size_t)(k0 + b_k) * N_DIM + (n0 + b_n);
    __builtin_prefetch(ap, 0, 1);       // -> global_prefetch_b8
    __builtin_prefetch(bp, 0, 1);
  };

  auto stage_to_lds = [&]() {
    // ---- A: row-major [m][k], 16 contiguous k per thread -> 2x b128 each
    const float* af = reinterpret_cast<const float*>(a4);
    v8bf h0, h1, l0, l1;
#pragma unroll
    for (int i = 0; i < 8; ++i) {
      float x  = af[i];
      __bf16 h = (__bf16)x;
      h0[i] = h;
      l0[i] = (__bf16)(x - (float)h);
    }
#pragma unroll
    for (int i = 0; i < 8; ++i) {
      float x  = af[8 + i];
      __bf16 h = (__bf16)x;
      h1[i] = h;
      l1[i] = (__bf16)(x - (float)h);
    }
    bf16_t* arow_hi = &Ahi[a_row * LDSS + a_col];
    bf16_t* arow_lo = &Alo[a_row * LDSS + a_col];
    *reinterpret_cast<v8bf*>(arow_hi)     = h0;
    *reinterpret_cast<v8bf*>(arow_hi + 8) = h1;
    *reinterpret_cast<v8bf*>(arow_lo)     = l0;
    *reinterpret_cast<v8bf*>(arow_lo + 8) = l1;

    // ---- B: transpose to [n][k]; pack the (k, k+1) pair -> 4-byte stores
    const float* bf_ = reinterpret_cast<const float*>(b4);
#pragma unroll
    for (int i = 0; i < 16; ++i) {
      float x0  = bf_[i];          // row k
      float x1  = bf_[16 + i];     // row k+1
      __bf16 hA = (__bf16)x0;
      __bf16 hB = (__bf16)x1;
      v2bf hp = {hA, hB};
      v2bf lp = {(__bf16)(x0 - (float)hA), (__bf16)(x1 - (float)hB)};
      const int off = (b_n + i) * LDSS + b_k;
      *reinterpret_cast<v2bf*>(&Bhi[off]) = hp;
      *reinterpret_cast<v2bf*>(&Blo[off]) = lp;
    }
  };

  // K-packed fragment load (identical pattern for A and transposed B)
  const int fr = lane & 15;
  const int fh = (lane >> 4) * 8;
  auto frag = [&](const bf16_t* base, int row) -> v16bf {
    const bf16_t* p = base + row * LDSS;
    v8bf x0 = *reinterpret_cast<const v8bf*>(p + fh);
    v8bf x1 = *reinterpret_cast<const v8bf*>(p + 16 + fh);
    return __builtin_shufflevector(x0, x1, 0, 1, 2, 3, 4, 5, 6, 7,
                                            8, 9, 10, 11, 12, 13, 14, 15);
  };

  load_step(0);
  prefetch_step(1);

  for (int s = 0; s < STEPS; ++s) {
    __syncthreads();                 // previous compute done reading LDS
    stage_to_lds();
    __syncthreads();                 // tiles visible to all waves

    if (s + 1 < STEPS) load_step(s + 1);      // latency hides under WMMA
    if (s + 2 < STEPS) prefetch_step(s + 2);

    v16bf ah[4], bh[4];
#pragma unroll
    for (int ms = 0; ms < 4; ++ms) ah[ms] = frag(Ahi, wm * 64 + ms * 16 + fr);
#pragma unroll
    for (int ns = 0; ns < 4; ++ns) bh[ns] = frag(Bhi, wn * 64 + ns * 16 + fr);

    // pass 1: hi * hi
#pragma unroll
    for (int ms = 0; ms < 4; ++ms)
#pragma unroll
      for (int ns = 0; ns < 4; ++ns)
        acc[ms][ns] = __builtin_amdgcn_wmma_f32_16x16x32_bf16(
            false, ah[ms], false, bh[ns], (short)0, acc[ms][ns], false, false);

    // pass 2: hi * lo
    {
      v16bf bl[4];
#pragma unroll
      for (int ns = 0; ns < 4; ++ns) bl[ns] = frag(Blo, wn * 64 + ns * 16 + fr);
#pragma unroll
      for (int ms = 0; ms < 4; ++ms)
#pragma unroll
        for (int ns = 0; ns < 4; ++ns)
          acc[ms][ns] = __builtin_amdgcn_wmma_f32_16x16x32_bf16(
              false, ah[ms], false, bl[ns], (short)0, acc[ms][ns], false, false);
    }

    // pass 3: lo * hi
    {
      v16bf al[4];
#pragma unroll
      for (int ms = 0; ms < 4; ++ms) al[ms] = frag(Alo, wm * 64 + ms * 16 + fr);
#pragma unroll
      for (int ms = 0; ms < 4; ++ms)
#pragma unroll
        for (int ns = 0; ns < 4; ++ns)
          acc[ms][ns] = __builtin_amdgcn_wmma_f32_16x16x32_bf16(
              false, al[ms], false, bh[ns], (short)0, acc[ms][ns], false, false);
    }
  }

  // Epilogue: C layout — VGPR g holds (M = g + 8*(lane>>4), N = lane&15)
  const int ch = (lane >> 4) * 8;
#pragma unroll
  for (int ms = 0; ms < 4; ++ms) {
#pragma unroll
    for (int ns = 0; ns < 4; ++ns) {
      const int row0 = m0 + wm * 64 + ms * 16 + ch;
      const int col  = n0 + wn * 64 + ns * 16 + (lane & 15);
#pragma unroll
      for (int g = 0; g < 8; ++g)
        Out[(size_t)(row0 + g) * N_DIM + col] = acc[ms][ns][g];
    }
  }
}

extern "C" void kernel_launch(void* const* d_in, const int* in_sizes, int n_in,
                              void* d_out, int out_size, void* d_ws, size_t ws_size,
                              hipStream_t stream) {
  const float* X = (const float*)d_in[0];   // [8, 8192, 1024] fp32
  const float* W = (const float*)d_in[1];   // [8, 1024, 8192] fp32
  float* Out = (float*)d_out;               // [8, 1024, 8192] == [8192, 8192]

  dim3 grid(N_DIM / BTN, M_DIM / BTM);      // 32 x 64 blocks
  dim3 block(256);                          // 8 wave32s
  gemm_rs_wmma_bf16x3<<<grid, block, 0, stream>>>(X, W, Out);
}
